// TextSNN_74113955660198
// MI455X (gfx1250) — matrix operand order repaired and checked by reading, not compiled, not run on hardware
//
#include <hip/hip_runtime.h>
#include <hip/hip_bf16.h>

// ---------------------------------------------------------------------------
// SNN: out[b,o] = sum_t LIF2( LIF1( x[b,t,:]@W1^T + b1 ) @ W2^T + b2 )
// B=256 T=32 F_IN=1024 H=4096 O=512, tau=2, v_th=1, hard reset.
//
// MI455X / gfx1250 (wave32, WMMA):
//  - bf16 WMMA (v_wmma_f32_16x16x32_bf16), f32 accumulate.
//  - M-tile = 32 rows = one batch element's 32 timesteps -> LIF scan fused
//    into the GEMM epilogue (per-wave LDS transpose). Y never materialized.
//  - Each wave computes a 32x32 tile: 4 WMMAs per 8 b128 loads (A and B
//    fragments each reused twice).
//  - Spikes stored as bf16 (exactly 0.0/1.0) -> GEMM2 input.
// ---------------------------------------------------------------------------

typedef __attribute__((ext_vector_type(16))) __bf16 v16bf;
typedef __attribute__((ext_vector_type(8)))  __bf16 v8bf;
typedef __attribute__((ext_vector_type(8)))  float  v8f;

union F16Frag { v16bf v; v8bf h[2]; };

#define SNN_B   256
#define SNN_T   32
#define SNN_FIN 1024
#define SNN_H   4096
#define SNN_O   512
#define SNN_M   (SNN_B * SNN_T)   // 8192 rows
#define LDS_LDA 33                // padded row stride (floats) -> no bank conflicts

// ---------------------------------------------------------------------------
// f32 -> bf16 (RNE), 4 elements per thread
// ---------------------------------------------------------------------------
__device__ __forceinline__ unsigned short snn_bf16(float f) {
    unsigned int u = __float_as_uint(f);
    return (unsigned short)((u + 0x7FFFu + ((u >> 16) & 1u)) >> 16);
}

__global__ __launch_bounds__(256) void snn_f32_to_bf16_v4(
    const float* __restrict__ in, unsigned short* __restrict__ out, int n4)
{
    int i = blockIdx.x * blockDim.x + threadIdx.x;
    if (i < n4) {
        float4 f = reinterpret_cast<const float4*>(in)[i];
        ushort4 o;
        o.x = snn_bf16(f.x); o.y = snn_bf16(f.y);
        o.z = snn_bf16(f.z); o.w = snn_bf16(f.w);
        reinterpret_cast<ushort4*>(out)[i] = o;
    }
}

// ---------------------------------------------------------------------------
// Core WMMA loop: one wave computes a 32(M) x 32(N) tile of A[M,K] @ B[N,K]^T.
//  A row-major [*, lda] bf16; B row-major [N, ldb] bf16 (weights are [N,K]).
//  A frag (16x32):  lane l: row m0+(l&15); K run  kh*16..kh*16+15 (contig)
//  B frag (32x16):  lane l: col n0+(l&15); K run  kh*16..kh*16+15 (contig)
// ---------------------------------------------------------------------------
template<int KDIM>
__device__ __forceinline__ void snn_gemm_body32(
    const unsigned short* __restrict__ A, int lda,
    const unsigned short* __restrict__ B, int ldb,
    int m0, int n0, int lane,
    v8f& c00, v8f& c01, v8f& c10, v8f& c11)
{
    const int khalf = lane >> 4;
    const int l15   = lane & 15;
    const __bf16* Ab = reinterpret_cast<const __bf16*>(A);
    const __bf16* Bb = reinterpret_cast<const __bf16*>(B);
    const __bf16* arow0 = Ab + (size_t)(m0 + l15) * lda;
    const __bf16* arow1 = arow0 + (size_t)16 * lda;
    const __bf16* brow0 = Bb + (size_t)(n0 + l15) * ldb + khalf * 16;
    const __bf16* brow1 = brow0 + (size_t)16 * ldb;

    #pragma unroll 2
    for (int kt = 0; kt < KDIM; kt += 32) {
        F16Frag a0, a1, b0, b1;
        a0.h[0] = *reinterpret_cast<const v8bf*>(arow0 + kt + khalf * 16);
        a0.h[1] = *reinterpret_cast<const v8bf*>(arow0 + kt + khalf * 16 + 8);
        a1.h[0] = *reinterpret_cast<const v8bf*>(arow1 + kt + khalf * 16);
        a1.h[1] = *reinterpret_cast<const v8bf*>(arow1 + kt + khalf * 16 + 8);
        b0.h[0] = *reinterpret_cast<const v8bf*>(brow0 + kt);
        b0.h[1] = *reinterpret_cast<const v8bf*>(brow0 + kt + 8);
        b1.h[0] = *reinterpret_cast<const v8bf*>(brow1 + kt);
        b1.h[1] = *reinterpret_cast<const v8bf*>(brow1 + kt + 8);

        if (kt + 32 < KDIM) {  // prefetch next K-tile of the streamed A rows
            __builtin_prefetch(arow0 + kt + 32, 0, 3);
            __builtin_prefetch(arow1 + kt + 32, 0, 3);
        }

        c00 = __builtin_amdgcn_wmma_f32_16x16x32_bf16(
            false, a0.v, false, b0.v, (short)0, c00, false, false);
        c10 = __builtin_amdgcn_wmma_f32_16x16x32_bf16(
            false, a1.v, false, b0.v, (short)0, c10, false, false);
        c01 = __builtin_amdgcn_wmma_f32_16x16x32_bf16(
            false, a0.v, false, b1.v, (short)0, c01, false, false);
        c11 = __builtin_amdgcn_wmma_f32_16x16x32_bf16(
            false, a1.v, false, b1.v, (short)0, c11, false, false);
    }
}

// Stage the four 16x16 f32 accumulators (lane-striped C layout) into a
// time-major [32 x 32] padded LDS tile for the sequential LIF scan.
__device__ __forceinline__ void snn_stage_acc32(
    float* __restrict__ ldsw, int lane,
    const v8f& c00, const v8f& c01, const v8f& c10, const v8f& c11,
    float bias0, float bias1)
{
    const int khalf = lane >> 4;
    const int l15   = lane & 15;
    #pragma unroll
    for (int i = 0; i < 8; ++i) {
        const int r = khalf * 8 + i;
        ldsw[r        * LDS_LDA + l15]      = c00[i] + bias0;
        ldsw[r        * LDS_LDA + 16 + l15] = c01[i] + bias1;
        ldsw[(16 + r) * LDS_LDA + l15]      = c10[i] + bias0;
        ldsw[(16 + r) * LDS_LDA + 16 + l15] = c11[i] + bias1;
    }
}

// ---------------------------------------------------------------------------
// GEMM1 + LIF1 -> bf16 spikes S1[8192, 4096]
// grid = (H/256, B) ; block = 256 (8 waves, each wave: 32x32 tile)
// ---------------------------------------------------------------------------
__global__ __launch_bounds__(256) void snn_gemm1_lif1(
    const unsigned short* __restrict__ Xb,   // [8192,1024] bf16
    const unsigned short* __restrict__ W1b,  // [4096,1024] bf16
    const float* __restrict__ b1,            // [4096]
    unsigned short* __restrict__ S1)         // [8192,4096] bf16 spikes
{
    __shared__ float lds[8][SNN_T * LDS_LDA];
    const int lane = threadIdx.x & 31;
    const int w    = threadIdx.x >> 5;
    const int n0   = blockIdx.x * 256 + w * 32;
    const int m0   = blockIdx.y * SNN_T;     // one batch element's timesteps

    v8f c00 = {}, c01 = {}, c10 = {}, c11 = {};
    snn_gemm_body32<SNN_FIN>(Xb, SNN_FIN, W1b, SNN_FIN, m0, n0, lane,
                             c00, c01, c10, c11);

    snn_stage_acc32(lds[w], lane, c00, c01, c10, c11,
                    b1[n0 + (lane & 15)], b1[n0 + 16 + (lane & 15)]);
    __syncthreads();

    // all 32 lanes: one lane per output feature, sequential LIF over T
    float v = 0.0f;
    size_t base = (size_t)m0 * SNN_H + n0 + lane;
    #pragma unroll
    for (int t = 0; t < SNN_T; ++t) {
        float y = lds[w][t * LDS_LDA + lane];
        v = 0.5f * (v + y);                  // v += (y - v)/tau, tau = 2
        unsigned short sp = 0;
        if (v >= 1.0f) { sp = 0x3F80u; v = 0.0f; }  // spike + hard reset
        S1[base + (size_t)t * SNN_H] = sp;
    }
}

// ---------------------------------------------------------------------------
// GEMM2 + LIF2 + sum_t -> out[256, 512] f32 spike counts
// grid = (O/256, B) ; block = 256
// ---------------------------------------------------------------------------
__global__ __launch_bounds__(256) void snn_gemm2_lif2(
    const unsigned short* __restrict__ S1,   // [8192,4096] bf16 spikes
    const unsigned short* __restrict__ W2b,  // [512,4096] bf16
    const float* __restrict__ b2,            // [512]
    float* __restrict__ out)                 // [256,512]
{
    __shared__ float lds[8][SNN_T * LDS_LDA];
    const int lane = threadIdx.x & 31;
    const int w    = threadIdx.x >> 5;
    const int n0   = blockIdx.x * 256 + w * 32;
    const int m0   = blockIdx.y * SNN_T;

    v8f c00 = {}, c01 = {}, c10 = {}, c11 = {};
    snn_gemm_body32<SNN_H>(S1, SNN_H, W2b, SNN_H, m0, n0, lane,
                           c00, c01, c10, c11);

    snn_stage_acc32(lds[w], lane, c00, c01, c10, c11,
                    b2[n0 + (lane & 15)], b2[n0 + 16 + (lane & 15)]);
    __syncthreads();

    float v = 0.0f, cnt = 0.0f;
    #pragma unroll
    for (int t = 0; t < SNN_T; ++t) {
        float z = lds[w][t * LDS_LDA + lane];
        v = 0.5f * (v + z);
        if (v >= 1.0f) { cnt += 1.0f; v = 0.0f; }
    }
    out[(size_t)blockIdx.y * SNN_O + n0 + lane] = cnt;
}

// ---------------------------------------------------------------------------
// Host launcher
// ---------------------------------------------------------------------------
extern "C" void kernel_launch(void* const* d_in, const int* in_sizes, int n_in,
                              void* d_out, int out_size, void* d_ws, size_t ws_size,
                              hipStream_t stream)
{
    (void)in_sizes; (void)n_in; (void)out_size; (void)ws_size;

    const float* x  = (const float*)d_in[0];   // [256,32,1024] == [8192,1024]
    const float* W1 = (const float*)d_in[1];   // [4096,1024]
    const float* b1 = (const float*)d_in[2];   // [4096]
    const float* W2 = (const float*)d_in[3];   // [512,4096]
    const float* b2 = (const float*)d_in[4];   // [512]
    float* out = (float*)d_out;                // [256,512]

    // Workspace layout (bf16 elements), total ~96.5 MB:
    const size_t xb_elems = (size_t)SNN_M * SNN_FIN;  //  8,388,608
    const size_t w1_elems = (size_t)SNN_H * SNN_FIN;  //  4,194,304
    const size_t w2_elems = (size_t)SNN_O * SNN_H;    //  2,097,152
    unsigned short* Xb  = (unsigned short*)d_ws;
    unsigned short* W1b = Xb  + xb_elems;
    unsigned short* W2b = W1b + w1_elems;
    unsigned short* S1  = W2b + w2_elems;             // 33,554,432 elems

    // 1) convert inputs to bf16 (4 elems/thread)
    {
        int n4 = (int)(xb_elems / 4);
        snn_f32_to_bf16_v4<<<(n4 + 255) / 256, 256, 0, stream>>>(x, Xb, n4);
        n4 = (int)(w1_elems / 4);
        snn_f32_to_bf16_v4<<<(n4 + 255) / 256, 256, 0, stream>>>(W1, W1b, n4);
        n4 = (int)(w2_elems / 4);
        snn_f32_to_bf16_v4<<<(n4 + 255) / 256, 256, 0, stream>>>(W2, W2b, n4);
    }

    // 2) GEMM1 + LIF1 -> spikes
    {
        dim3 grid(SNN_H / 256, SNN_B);
        snn_gemm1_lif1<<<grid, 256, 0, stream>>>(Xb, W1b, b1, S1);
    }

    // 3) GEMM2 + LIF2 + temporal sum -> out
    {
        dim3 grid(SNN_O / 256, SNN_B);
        snn_gemm2_lif2<<<grid, 256, 0, stream>>>(S1, W2b, b2, out);
    }
}
